// MultiLayerGAT_74431783240149
// MI455X (gfx1250) — matrix operand order repaired and checked by reading, not compiled, not run on hardware
//
#include <hip/hip_runtime.h>

typedef __attribute__((ext_vector_type(16))) _Float16 v16h;
typedef __attribute__((ext_vector_type(8)))  float    v8f;

#define N_NODES 50000
#define E_EDGES 800000
#define IN_DIM  128
#define HID     16
#define HEADS   4
#define HC      64
#define OUT_DIM 64
#define NEG_SLOPE 0.2f

// ---- order-preserving float <-> uint encoding for atomic segment-max ----
__device__ __forceinline__ unsigned encOrd(float f) {
    unsigned b = __float_as_uint(f);
    return b ^ (unsigned)(((int)b >> 31) | 0x80000000);
}
__device__ __forceinline__ float decOrd(unsigned u) {
    if (u == 0u) return 0.0f;  // no incoming edge -> m := 0 (matches isfinite guard)
    unsigned b = (u & 0x80000000u) ? (u ^ 0x80000000u) : ~u;
    return __uint_as_float(b);
}
__device__ __forceinline__ void atomAddF(float* p, float v) {
#if defined(__HIP_PLATFORM_AMD__)
    unsafeAtomicAdd(p, v);   // global_atomic_add_f32; targets stay L2-resident
#else
    atomicAdd(p, v);
#endif
}

__device__ __forceinline__ void cvt4(v16h& a, int base, const float4 f) {
    a[base + 0] = (_Float16)f.x;
    a[base + 1] = (_Float16)f.y;
    a[base + 2] = (_Float16)f.z;
    a[base + 3] = (_Float16)f.w;
}

// ---- pack W[Kdim, outW] (f32, row-major) into WMMA B-lane layout (f16) ----
// pB[((kc*ntiles + tileN)*32 + lane)*16 + e] = W[kc*32 + (lane/16)*8 + (e < 8 ? e : 8 + e)][tileN*16 + lane%16]
// Zero-padded for k >= Kdim. Tiny (<= 40 KB total), one small launch per weight.
__global__ void pack_w_kernel(const float* __restrict__ W, int Kdim, int outW,
                              _Float16* __restrict__ pB)
{
    const int kchunks = (Kdim + 31) >> 5;
    const int ntiles  = outW >> 4;
    const int total   = kchunks * ntiles * 32;
    int t = blockIdx.x * blockDim.x + threadIdx.x;
    if (t >= total) return;
    const int lane = t & 31;
    const int tile = (t >> 5) % ntiles;
    const int kc   = (t >> 5) / ntiles;
    const int col  = tile * 16 + (lane & 15);
    const int koff = (lane >> 4) << 3;
    _Float16* dst = pB + (size_t)t * 16;
#pragma unroll
    for (int e = 0; e < 16; ++e) {
        int k = (kc << 5) + koff + (e < 8 ? e : 8 + e);
        float v = (k < Kdim) ? W[(size_t)k * outW + col] : 0.0f;
        dst[e] = (_Float16)v;
    }
}

// ---- WMMA GEMM: out[N_NODES, OUTW] = A[N_NODES, KDIM] @ W (+bias) ----
// KDIM/OUTW compile-time: A fragment = 2..4 x b128 loads, B fragment = one v16h load.
template<int KDIM, int OUTW, bool HAS_BIAS>
__global__ void gemm_wmma(const float* __restrict__ A,
                          const _Float16* __restrict__ pB,
                          const float* __restrict__ bias,
                          float* __restrict__ out)
{
    constexpr int NTILES  = OUTW / 16;
    constexpr int MTILES  = N_NODES / 16;       // 3125, exact (no M tail)
    constexpr int KCHUNKS = (KDIM + 31) / 32;

    const int lane = threadIdx.x & 31;
    const int wave = blockIdx.x * (blockDim.x >> 5) + (threadIdx.x >> 5);
    if (wave >= MTILES * NTILES) return;        // wave-uniform: EXEC all-ones inside
    const int tileM = wave / NTILES;
    const int tileN = wave % NTILES;

    const int mrow  = tileM * 16 + (lane & 15);
    const int koffA = (lane >> 4) << 3;         // 0 or 8
    const float*    Arow = A + (size_t)mrow * KDIM + koffA;
    const _Float16* Bp   = pB + ((size_t)tileN * 32 + lane) * 16;

    v8f c = {};
#pragma unroll
    for (int kc = 0; kc < KCHUNKS; ++kc) {
        const float* p0 = Arow + kc * 32;
        v16h a;
        cvt4(a, 0, *(const float4*)(p0));           // K = kbase + koffA + 0..3
        cvt4(a, 4, *(const float4*)(p0 + 4));       // K = kbase + koffA + 4..7
        if constexpr (KDIM % 32 == 0) {
            cvt4(a, 8,  *(const float4*)(p0 + 16)); // K = kbase + 16 + koffA + 0..3
            cvt4(a, 12, *(const float4*)(p0 + 20));
        } else {                                    // KDIM==16: upper half zero-padded
#pragma unroll
            for (int e = 8; e < 16; ++e) a[e] = (_Float16)0.0f;
        }
        const v16h b = *(const v16h*)(Bp + (size_t)kc * NTILES * 32 * 16);
        c = __builtin_amdgcn_wmma_f32_16x16x32_f16(false, a, false, b,
                                                   (short)0, c, false, false);
    }

    const int row0 = tileM * 16 + ((lane >> 4) << 3);
    const int col  = tileN * 16 + (lane & 15);
    const float bv = HAS_BIAS ? bias[col] : 0.0f;
    float* orow = out + (size_t)row0 * OUTW + col;
#pragma unroll
    for (int r = 0; r < 8; ++r)
        orow[(size_t)r * OUTW] = c[r] + bv;
}

// ---- per-node attention logits: alpha_s/d[n,h] = sum_c xt[n,h,c] * a[h,c] ----
__global__ void alpha_kernel(const float* __restrict__ xt,
                             const float* __restrict__ a_src,
                             const float* __restrict__ a_dst,
                             float* __restrict__ alpha_s,
                             float* __restrict__ alpha_d)
{
    int i = blockIdx.x * blockDim.x + threadIdx.x;
    if (i >= N_NODES * HEADS) return;
    int n = i >> 2, h = i & 3;
    const float* xr = xt + (size_t)n * HC + h * HID;
    const float* as = a_src + h * HID;
    const float* ad = a_dst + h * HID;
    float ss = 0.0f, sd = 0.0f;
#pragma unroll
    for (int c = 0; c < HID; ++c) { float v = xr[c]; ss += v * as[c]; sd += v * ad[c]; }
    alpha_s[i] = ss;
    alpha_d[i] = sd;
}

__device__ __forceinline__ float edge_logit(const float* __restrict__ alpha_s,
                                            const float* __restrict__ alpha_d,
                                            long long s, long long d, int h)
{
    float v = alpha_s[s * HEADS + h] + alpha_d[d * HEADS + h];
    return v > 0.0f ? v : NEG_SLOPE * v;
}

// ---- pass 1: segment max over incoming edges (atomic max, ordered-uint) ----
__global__ void edge_max_kernel(const long long* __restrict__ src,
                                const long long* __restrict__ dst,
                                const float* __restrict__ alpha_s,
                                const float* __restrict__ alpha_d,
                                unsigned* __restrict__ m_u)
{
    int i = blockIdx.x * blockDim.x + threadIdx.x;
    if (i >= E_EDGES * HEADS) return;
    int e = i >> 2, h = i & 3;
    long long s = src[e], d = dst[e];
    float v = edge_logit(alpha_s, alpha_d, s, d, h);
    atomicMax(&m_u[d * HEADS + h], encOrd(v));
}

// ---- pass 2: segment sum of exp(e - m) ----
__global__ void edge_den_kernel(const long long* __restrict__ src,
                                const long long* __restrict__ dst,
                                const float* __restrict__ alpha_s,
                                const float* __restrict__ alpha_d,
                                const unsigned* __restrict__ m_u,
                                float* __restrict__ denom)
{
    int i = blockIdx.x * blockDim.x + threadIdx.x;
    if (i >= E_EDGES * HEADS) return;
    int e = i >> 2, h = i & 3;
    long long s = src[e], d = dst[e];
    float v = edge_logit(alpha_s, alpha_d, s, d, h);
    float m = decOrd(m_u[d * HEADS + h]);
    atomAddF(&denom[d * HEADS + h], __expf(v - m));
}

// ---- pass 3: weighted message aggregation (64 lanes per edge, coalesced) ----
__global__ void edge_agg_kernel(const long long* __restrict__ src,
                                const long long* __restrict__ dst,
                                const float* __restrict__ alpha_s,
                                const float* __restrict__ alpha_d,
                                const unsigned* __restrict__ m_u,
                                const float* __restrict__ denom,
                                const float* __restrict__ xt,
                                float* __restrict__ agg)
{
    int t = blockIdx.x * blockDim.x + threadIdx.x;
    int e = t >> 6;
    if (e >= E_EDGES) return;
    int ch = t & 63, h = ch >> 4;
    long long s = src[e], d = dst[e];
    float v  = edge_logit(alpha_s, alpha_d, s, d, h);
    float m  = decOrd(m_u[d * HEADS + h]);
    float ee = __expf(v - m);
    float a  = ee / (denom[d * HEADS + h] + 1e-16f);
    atomAddF(&agg[(size_t)d * HC + ch], xt[(size_t)s * HC + ch] * a);
}

// ---- bias + ELU ----
__global__ void bias_elu_kernel(const float* __restrict__ agg,
                                const float* __restrict__ bias,
                                float* __restrict__ out)
{
    int i = blockIdx.x * blockDim.x + threadIdx.x;
    if (i >= N_NODES * HC) return;
    float v = agg[i] + bias[i & 63];
    out[i] = v > 0.0f ? v : (__expf(v) - 1.0f);
}

__global__ void zero_kernel(float* __restrict__ p, int n)
{
    int i = blockIdx.x * blockDim.x + threadIdx.x;
    if (i < n) p[i] = 0.0f;
}

static inline int cdiv(long long a, long long b) { return (int)((a + b - 1) / b); }

extern "C" void kernel_launch(void* const* d_in, const int* in_sizes, int n_in,
                              void* d_out, int out_size, void* d_ws, size_t ws_size,
                              hipStream_t stream)
{
    const float*     x       = (const float*)d_in[0];
    const long long* ei      = (const long long*)d_in[1];   // int64 per reference
    const long long* src     = ei;
    const long long* dstp    = ei + E_EDGES;
    const float*     proj_w  = (const float*)d_in[2];
    const float*     proj_b  = (const float*)d_in[3];
    const float*     lin_w0  = (const float*)d_in[4];
    const float*     lin_wr  = (const float*)d_in[5];
    const float*     att_src = (const float*)d_in[6];
    const float*     att_dst = (const float*)d_in[7];
    const float*     bias_l  = (const float*)d_in[8];
    const float*     out_w   = (const float*)d_in[9];
    const float*     out_b   = (const float*)d_in[10];
    float*           out     = (float*)d_out;

    // ---- workspace layout (~41.7 MB) ----
    float*    bufA    = (float*)d_ws;                           // [N,64] layer input
    float*    bufXT   = bufA   + (size_t)N_NODES * HC;          // [N,64] xt
    float*    bufAGG  = bufXT  + (size_t)N_NODES * HC;          // [N,64] aggregation
    float*    alpha_s = bufAGG + (size_t)N_NODES * HC;          // [N,4]
    float*    alpha_d = alpha_s + N_NODES * HEADS;              // [N,4]
    unsigned* m_u     = (unsigned*)(alpha_d + N_NODES * HEADS); // [N,4] ordered-uint max
    float*    denom   = (float*)(m_u + N_NODES * HEADS);        // [N,4]
    // packed f16 weights (B-lane layout), 32B-aligned (offset is multiple of 32)
    _Float16* pProj = (_Float16*)(denom + N_NODES * HEADS);     // 4 chunks * 1 tile * 512
    _Float16* pLin0 = pProj + 4 * 1 * 512;                      // 1 chunk  * 4 tiles
    _Float16* pLin1 = pLin0 + 1 * 4 * 512;                      // 2 chunks * 4 tiles each
    _Float16* pLin2 = pLin1 + 2 * 4 * 512;
    _Float16* pLin3 = pLin2 + 2 * 4 * 512;
    _Float16* pOutW = pLin3 + 2 * 4 * 512;

    // ---- pack all weights to WMMA B layout (tiny) ----
    pack_w_kernel<<<1, 128, 0, stream>>>(proj_w, IN_DIM, HID, pProj);
    pack_w_kernel<<<1, 128, 0, stream>>>(lin_w0, HID, HC, pLin0);
    pack_w_kernel<<<1, 256, 0, stream>>>(lin_wr + 0 * HC * HC, HC, HC, pLin1);
    pack_w_kernel<<<1, 256, 0, stream>>>(lin_wr + 1 * HC * HC, HC, HC, pLin2);
    pack_w_kernel<<<1, 256, 0, stream>>>(lin_wr + 2 * HC * HC, HC, HC, pLin3);
    pack_w_kernel<<<1, 256, 0, stream>>>(out_w, HC, OUT_DIM, pOutW);

    // proj: [N,128] @ [128,16] + b -> bufA (row stride 16)
    gemm_wmma<IN_DIM, HID, true><<<cdiv((N_NODES / 16) * 1, 8), 256, 0, stream>>>(
        x, pProj, proj_b, bufA);

    _Float16* pLin[4] = { pLin0, pLin1, pLin2, pLin3 };
    for (int l = 0; l < 4; ++l) {
        // xt = h @ W -> bufXT [N,64]
        if (l == 0)
            gemm_wmma<HID, HC, false><<<cdiv((N_NODES / 16) * 4, 8), 256, 0, stream>>>(
                bufA, pLin[l], nullptr, bufXT);
        else
            gemm_wmma<HC, HC, false><<<cdiv((N_NODES / 16) * 4, 8), 256, 0, stream>>>(
                bufA, pLin[l], nullptr, bufXT);

        alpha_kernel<<<cdiv(N_NODES * HEADS, 256), 256, 0, stream>>>(
            bufXT, att_src + l * HEADS * HID, att_dst + l * HEADS * HID,
            alpha_s, alpha_d);

        // zero m_u + denom (contiguous) and agg
        zero_kernel<<<cdiv(2 * N_NODES * HEADS, 256), 256, 0, stream>>>(
            (float*)m_u, 2 * N_NODES * HEADS);
        zero_kernel<<<cdiv(N_NODES * HC, 256), 256, 0, stream>>>(
            bufAGG, N_NODES * HC);

        edge_max_kernel<<<cdiv((long long)E_EDGES * HEADS, 256), 256, 0, stream>>>(
            src, dstp, alpha_s, alpha_d, m_u);
        edge_den_kernel<<<cdiv((long long)E_EDGES * HEADS, 256), 256, 0, stream>>>(
            src, dstp, alpha_s, alpha_d, m_u, denom);
        edge_agg_kernel<<<cdiv((long long)E_EDGES * 64, 256), 256, 0, stream>>>(
            src, dstp, alpha_s, alpha_d, m_u, denom, bufXT, bufAGG);

        bias_elu_kernel<<<cdiv(N_NODES * HC, 256), 256, 0, stream>>>(
            bufAGG, bias_l + l * HC, bufA);
    }

    // out = h @ out_w + out_b -> d_out [N,64]
    gemm_wmma<HC, OUT_DIM, true><<<cdiv((N_NODES / 16) * 4, 8), 256, 0, stream>>>(
        bufA, pOutW, out_b, out);
}
